// SelfAttention_22789096472642
// MI455X (gfx1250) — matrix operand order repaired and checked by reading, not compiled, not run on hardware
//
#include <hip/hip_runtime.h>
#include <stdint.h>

// ---------------------------------------------------------------------------
// Self-attention for MI455X (gfx1250, wave32, WMMA).
// x:[8,2048,512] f32, kernel:[3,512,512] f32 -> out:[8,2048,512] f32.
// Compute-bound (~94 GFLOP vs ~67 MB HBM traffic) -> bf16 WMMA path:
//   v_wmma_f32_16x16x32_bf16 (D = A(16x32 bf16) x B(32x16 bf16) + C(f32)).
// ---------------------------------------------------------------------------

typedef __attribute__((ext_vector_type(16))) __bf16 v16bf;
typedef __attribute__((ext_vector_type(8)))  float  v8f;

union FragBF {
    v16bf    bf;
    uint32_t u[8];   // 8 dwords = 16 packed bf16 per lane
};

__device__ __forceinline__ unsigned short f32_to_bf16(float f) {
    union { float f; uint32_t u; } cv; cv.f = f;
    uint32_t u = cv.u;
    u += 0x7fffu + ((u >> 16) & 1u);   // round-to-nearest-even
    return (unsigned short)(u >> 16);
}

#define B_   8
#define S_   2048
#define D_   512
#define M_   (B_ * S_)          // 16384 flattened (b,s) rows
#define SCALE_ 0.125f           // 1/sqrt(64)

// ---------------------------------------------------------------------------
// Kernel 1: precision prep. x -> bf16 (row-major), W -> W^T bf16 (per qkv).
// Wt[qkv][e][d] = W[qkv][d][e]  so GEMM B-fragments are contiguous K-pairs.
// ---------------------------------------------------------------------------
__global__ __launch_bounds__(256) void prep_kernel(
        const float* __restrict__ x, const float* __restrict__ kern,
        unsigned short* __restrict__ xbf, unsigned short* __restrict__ Wt) {
    const uint32_t i = blockIdx.x * 256u + threadIdx.x;
    const uint32_t nx = (uint32_t)M_ * D_;          // 8,388,608
    if (i < nx) xbf[i] = f32_to_bf16(x[i]);
    const uint32_t nw = 3u * D_ * D_;               // 786,432
    if (i < nw) {
        uint32_t qkv = i / (D_ * D_);
        uint32_t rem = i - qkv * (D_ * D_);
        uint32_t e = rem / D_, d = rem - e * D_;
        Wt[i] = f32_to_bf16(kern[qkv * (D_ * D_) + d * D_ + e]);
    }
}

// ---------------------------------------------------------------------------
// Kernel 2: QKV projection GEMM. 8 waves/WG, 16x16 f32 acc per wave,
// K-loop: 16 x v_wmma_f32_16x16x32_bf16.  grid = (N/32, M/64, 3).
// Q,K stored row-major bf16; V stored transposed Vt[b][e][s] bf16 so that
// the attention P@V GEMM can load B fragments with contiguous pairs.
// ---------------------------------------------------------------------------
__global__ __launch_bounds__(256) void qkv_gemm_kernel(
        const unsigned short* __restrict__ xbf,
        const unsigned short* __restrict__ Wt,
        unsigned short* __restrict__ Q,
        unsigned short* __restrict__ K,
        unsigned short* __restrict__ Vt) {
    const int lane = threadIdx.x & 31;
    const int w    = threadIdx.x >> 5;
    const int row0 = blockIdx.y * 64 + (w & 3) * 16;
    const int col0 = blockIdx.x * 32 + (w >> 2) * 16;
    const int qkv  = blockIdx.z;
    const int hi   = lane >> 4;
    const int l15  = lane & 15;

    const unsigned short* arow = xbf + (size_t)(row0 + l15) * D_;
    const unsigned short* brow = Wt + (size_t)qkv * D_ * D_ + (size_t)(col0 + l15) * D_;

    v8f c = {};
    for (int k0 = 0; k0 < D_; k0 += 32) {
        FragBF a, b;
#pragma unroll
        for (int v = 0; v < 8; ++v) {
            const int ka = k0 + ((v >> 2) << 4) + (hi << 3) + ((v & 3) << 1);
            a.u[v] = *(const uint32_t*)(arow + ka);          // A: 16x32 bf16 layout
            const int kb = k0 + ((v >> 2) << 4) + (hi << 3) + ((v & 3) << 1);
            b.u[v] = *(const uint32_t*)(brow + kb);          // B: 32x16 bf16 layout
        }
        c = __builtin_amdgcn_wmma_f32_16x16x32_bf16(false, a.bf, false, b.bf,
                                                    (short)0, c, false, false);
    }

    const int n  = l15;
    const int mb = hi << 3;
    if (qkv < 2) {
        unsigned short* dst = (qkv == 0) ? Q : K;
#pragma unroll
        for (int r = 0; r < 8; ++r)
            dst[(size_t)(row0 + mb + r) * D_ + col0 + n] = f32_to_bf16(c[r]);
    } else {
#pragma unroll
        for (int r = 0; r < 8; ++r) {
            const int m = row0 + mb + r;
            const int bidx = m >> 11, s = m & (S_ - 1);
            Vt[((size_t)bidx * D_ + (col0 + n)) * S_ + s] = f32_to_bf16(c[r]);
        }
    }
}

// ---------------------------------------------------------------------------
// Kernel 3: flash attention. One WG (8 waves) per (batch, 16-row q-block).
// kv loop in blocks of 128: S=Q@K^T (each wave one 16x16 tile, K-dim 512),
// online softmax via LDS, P(bf16) in LDS, O += P@V (each wave 16x64 of O).
// ---------------------------------------------------------------------------
__global__ __launch_bounds__(256) void flash_attn_kernel(
        const unsigned short* __restrict__ Q,
        const unsigned short* __restrict__ K,
        const unsigned short* __restrict__ Vt,
        float* __restrict__ out) {
    __shared__ float          sS[16][128];
    __shared__ unsigned short sP[16][128];
    __shared__ float          red[16][16];
    __shared__ float          rowmax[16], rowsum[16], rowscale[16];

    const int lane = threadIdx.x & 31;
    const int w    = threadIdx.x >> 5;
    const int b    = blockIdx.y;
    const int q0   = blockIdx.x * 16;
    const int hi   = lane >> 4;
    const int l15  = lane & 15;

    if (threadIdx.x < 16) { rowmax[threadIdx.x] = -1e30f; rowsum[threadIdx.x] = 0.f; }

    v8f o0 = {}, o1 = {}, o2 = {}, o3 = {};
    v8f* o[4] = { &o0, &o1, &o2, &o3 };

    const unsigned short* qrow = Q + (size_t)(b * S_ + q0 + l15) * D_;
    __syncthreads();

    for (int kv0 = 0; kv0 < S_; kv0 += 128) {
        // ---- stage 1: S tile = Q(16x512) x K^T(512x16), one tile per wave
        v8f s = {};
        const unsigned short* krow = K + (size_t)(b * S_ + kv0 + w * 16 + l15) * D_;
        for (int k0 = 0; k0 < D_; k0 += 32) {
            FragBF a, bb;
#pragma unroll
            for (int v = 0; v < 8; ++v) {
                const int ka = k0 + ((v >> 2) << 4) + (hi << 3) + ((v & 3) << 1);
                a.u[v]  = *(const uint32_t*)(qrow + ka);
                const int kb = k0 + ((v >> 2) << 4) + (hi << 3) + ((v & 3) << 1);
                bb.u[v] = *(const uint32_t*)(krow + kb);   // B = K^T: rows of K
            }
            s = __builtin_amdgcn_wmma_f32_16x16x32_bf16(false, a.bf, false, bb.bf,
                                                        (short)0, s, false, false);
        }
#pragma unroll
        for (int r = 0; r < 8; ++r)
            sS[(hi << 3) + r][w * 16 + l15] = s[r] * SCALE_;
        __syncthreads();

        // ---- online softmax: block max per row (16 threads x 8 cols each)
        {
            const int row = threadIdx.x >> 4, seg = threadIdx.x & 15;
            float m = -1e30f;
#pragma unroll
            for (int cc = 0; cc < 8; ++cc) m = fmaxf(m, sS[row][seg * 8 + cc]);
            red[row][seg] = m;
        }
        __syncthreads();
        if (threadIdx.x < 16) {
            const int row = threadIdx.x;
            float bm = red[row][0];
#pragma unroll
            for (int i = 1; i < 16; ++i) bm = fmaxf(bm, red[row][i]);
            const float mo = rowmax[row];
            const float mn = fmaxf(mo, bm);
            const float sc = __expf(mo - mn);
            rowscale[row] = sc;
            rowmax[row]   = mn;
            rowsum[row]  *= sc;
        }
        __syncthreads();
        {
            const int row = threadIdx.x >> 4, seg = threadIdx.x & 15;
            const float mrow = rowmax[row];
            float ps = 0.f;
#pragma unroll
            for (int cc = 0; cc < 8; ++cc) {
                const int c = seg * 8 + cc;
                const float p = __expf(sS[row][c] - mrow);
                sP[row][c] = f32_to_bf16(p);
                ps += p;
            }
            red[row][seg] = ps;
        }
        __syncthreads();
        if (threadIdx.x < 16) {
            const int row = threadIdx.x;
            float ssum = 0.f;
#pragma unroll
            for (int i = 0; i < 16; ++i) ssum += red[row][i];
            rowsum[row] += ssum;
        }

        // ---- stage 2: O(16x64 per wave) = rescale(O) + P(16x128) x V(128x64)
        float sc[8];
#pragma unroll
        for (int r = 0; r < 8; ++r) sc[r] = rowscale[(hi << 3) + r];
#pragma unroll
        for (int t = 0; t < 4; ++t)
#pragma unroll
            for (int r = 0; r < 8; ++r) (*o[t])[r] *= sc[r];

        for (int ks = 0; ks < 4; ++ks) {              // K-dim 128 = 4 x 32
            FragBF a;
#pragma unroll
            for (int v = 0; v < 8; ++v) {
                const int ka = ks * 32 + ((v >> 2) << 4) + (hi << 3) + ((v & 3) << 1);
                a.u[v] = *(const uint32_t*)(&sP[l15][ka]);      // ds_load_b32
            }
#pragma unroll
            for (int t = 0; t < 4; ++t) {
                const int col0 = w * 64 + t * 16;
                const unsigned short* vrow =
                    Vt + ((size_t)b * D_ + (col0 + l15)) * S_ + kv0 + ks * 32;
                FragBF bb;
#pragma unroll
                for (int v = 0; v < 8; ++v)
                    bb.u[v] = *(const uint32_t*)(vrow + ((v >> 2) << 4) + (hi << 3) + ((v & 3) << 1));
                *o[t] = __builtin_amdgcn_wmma_f32_16x16x32_bf16(false, a.bf, false, bb.bf,
                                                                (short)0, *o[t], false, false);
            }
        }
        __syncthreads();   // protect sP/rowscale before next kv block overwrites
    }

    // ---- finalize: divide by row sums, store f32
    float invs[8];
#pragma unroll
    for (int r = 0; r < 8; ++r) {
        const float rs = rowsum[(hi << 3) + r];
        invs[r] = (rs > 0.f) ? (1.0f / rs) : 0.f;
    }
#pragma unroll
    for (int t = 0; t < 4; ++t) {
        const int col0 = w * 64 + t * 16;
#pragma unroll
        for (int r = 0; r < 8; ++r) {
            const int m = (hi << 3) + r;
            out[(size_t)(b * S_ + q0 + m) * D_ + col0 + l15] = (*o[t])[r] * invs[r];
        }
    }
}

// ---------------------------------------------------------------------------
// Workspace layout (bytes):
//   xbf : 0        .. 16 MiB   (16384x512 bf16)
//   Wt  : 16 MiB   .. +1.5 MiB (3x512x512 bf16, transposed)
//   Q   : +16 MiB
//   K   : +16 MiB
//   Vt  : +16 MiB  ([b][d][s] bf16)
// ---------------------------------------------------------------------------
extern "C" void kernel_launch(void* const* d_in, const int* in_sizes, int n_in,
                              void* d_out, int out_size, void* d_ws, size_t ws_size,
                              hipStream_t stream) {
    const float* x    = (const float*)d_in[0];
    const float* kern = (const float*)d_in[1];
    float*       out  = (float*)d_out;

    char* ws = (char*)d_ws;
    unsigned short* xbf = (unsigned short*)(ws);
    unsigned short* Wt  = (unsigned short*)(ws + (size_t)16777216);
    unsigned short* Q   = (unsigned short*)(ws + (size_t)18350080);
    unsigned short* K   = (unsigned short*)(ws + (size_t)35127296);
    unsigned short* Vt  = (unsigned short*)(ws + (size_t)51904512);

    // 1) fp32 -> bf16 staging (x row-major, W transposed)
    prep_kernel<<<32768, 256, 0, stream>>>(x, kern, xbf, Wt);

    // 2) Q/K/V projections: grid = (N/32, M/64, qkv)
    qkv_gemm_kernel<<<dim3(16, 256, 3), 256, 0, stream>>>(xbf, Wt, Q, K, Vt);

    // 3) flash attention: grid = (q-blocks, batches)
    flash_attn_kernel<<<dim3(S_ / 16, B_), 256, 0, stream>>>(Q, K, Vt, out);
}
